// Model_27762668601950
// MI455X (gfx1250) — matrix operand order repaired and checked by reading, not compiled, not run on hardware
//
#include <hip/hip_runtime.h>

typedef __attribute__((ext_vector_type(16))) __bf16 v16bf;
typedef __attribute__((ext_vector_type(8)))  float  v8f;

#define WAVES 4
#define TPB   128

__device__ __forceinline__ unsigned short f2bf(float f){
  union { float f; unsigned u; } v; v.f = f;
  unsigned u = v.u;
  return (unsigned short)((u + 0x7FFFu + ((u >> 16) & 1u)) >> 16);
}
__device__ __forceinline__ float bf2f(unsigned short h){
  union { unsigned u; float f; } v; v.u = ((unsigned)h) << 16; return v.f;
}

union ABFrag { v16bf v; uint4 q[2]; };

// A fragment (16x32 bf16, M x K) from row-major LDS [16][stride]
// ISA layout: lanes 0-15 rows M, half=lane>>4; elems 0-7 -> K=k0+8*half.., elems 8-15 -> K=k0+16+8*half..
__device__ __forceinline__ v16bf load_A(const unsigned short* base, int stride, int k0){
  const int lane = threadIdx.x & 31;
  const int m = lane & 15, half = lane >> 4;
  ABFrag r;
  r.q[0] = *(const uint4*)(base + m*stride + k0 + 8*half);
  r.q[1] = *(const uint4*)(base + m*stride + k0 + 16 + 8*half);
  return r.v;
}

// B fragment from pre-packed weights: 32 lanes x 16 bf16 contiguous per tile
__device__ __forceinline__ v16bf load_B(const unsigned short* packed, int tile){
  const int lane = threadIdx.x & 31;
  const uint4* p = (const uint4*)(packed + ((size_t)tile*32 + lane)*16);
  ABFrag r; r.q[0] = p[0]; r.q[1] = p[1];
  return r.v;
}

// D fragment (16x16 f32): VGPR r -> M=r+8*half, N=lane&15
__device__ __forceinline__ void store_D(float* G, int gstride, int n0, v8f acc){
  const int lane = threadIdx.x & 31;
  const int n = lane & 15, half = lane >> 4;
  #pragma unroll
  for (int r = 0; r < 8; r++)
    G[(r + half*8)*gstride + n0 + n] = acc[r];
}

// G[e] = [P_e | Q_e] @ W (bf16 WMMA, f32 accum) for two batch elems sharing B fragments
__device__ __forceinline__ void mm2(
    const unsigned short* P0, const unsigned short* P1, int pstride, int KTp,
    const unsigned short* Q0, const unsigned short* Q1, int qstride, int KTq,
    const unsigned short* Wp, int NT, float* G0, float* G1, int gstride)
{
  const int wave = threadIdx.x >> 5;
  const int KT = KTp + KTq;
  for (int nt = wave; nt < NT; nt += WAVES){
    v8f acc0 = {}; v8f acc1 = {};
    for (int kt = 0; kt < KT; kt++){
      v16bf bf = load_B(Wp, nt*KT + kt);
      v16bf a0, a1;
      if (kt < KTp){ a0 = load_A(P0, pstride, kt*32);        a1 = load_A(P1, pstride, kt*32); }
      else         { a0 = load_A(Q0, qstride, (kt-KTp)*32);  a1 = load_A(Q1, qstride, (kt-KTp)*32); }
      acc0 = __builtin_amdgcn_wmma_f32_16x16x32_bf16(false, a0, false, bf, (short)0, acc0, false, false);
      acc1 = __builtin_amdgcn_wmma_f32_16x16x32_bf16(false, a1, false, bf, (short)0, acc1, false, false);
    }
    store_D(G0, gstride, nt*16, acc0);
    store_D(G1, gstride, nt*16, acc1);
  }
}

// ru = sigmoid(L @ G + bias); r*h -> RH (bf16), u -> U (bf16)
__device__ __forceinline__ void gates_ew(const float* L, const float* G,
                                         const unsigned short* H, unsigned short* RH,
                                         unsigned short* U, const float* __restrict__ bias)
{
  for (int idx = threadIdx.x; idx < 16*256; idx += TPB){
    const int i = idx >> 8, c = idx & 255;
    float s = bias[c];
    #pragma unroll
    for (int j = 0; j < 16; j++) s += L[i*16+j] * G[j*256+c];
    const float sig = 1.f/(1.f + expf(-s));
    if (c < 128) RH[i*128 + c]        = f2bf(sig * bf2f(H[i*128 + c]));
    else         U [i*128 + (c-128)]  = f2bf(sig);
  }
}

// c = tanh(L @ G); h = u*h + (1-u)*c   (in place on H)
__device__ __forceinline__ void cand_ew(const float* L, const float* G,
                                        const unsigned short* U, unsigned short* H)
{
  for (int idx = threadIdx.x; idx < 16*128; idx += TPB){
    const int i = idx >> 7, c = idx & 127;
    float s = 0.f;
    #pragma unroll
    for (int j = 0; j < 16; j++) s += L[i*16+j] * G[j*128+c];
    const float cc = tanhf(s);
    const float u = bf2f(U[idx]);
    const float h = bf2f(H[idx]);
    H[idx] = f2bf(u*h + (1.f - u)*cc);
  }
}

__global__ __launch_bounds__(TPB)
void tgcn_kernel(const float* __restrict__ feats, const float* __restrict__ Lg,
                 const unsigned short* __restrict__ pW1g, const float* __restrict__ b1g,
                 const unsigned short* __restrict__ pW1c,
                 const unsigned short* __restrict__ pW2g, const float* __restrict__ b2g,
                 const unsigned short* __restrict__ pW2c,
                 const float* __restrict__ bng, const float* __restrict__ bnb,
                 const float* __restrict__ bnm, const float* __restrict__ bnv,
                 const float* __restrict__ fcw, const float* __restrict__ fcb,
                 float* __restrict__ out)
{
  __shared__ __attribute__((aligned(16))) unsigned short sX [2][16*256];
  __shared__ __attribute__((aligned(16))) unsigned short sH1[2][16*128];
  __shared__ __attribute__((aligned(16))) unsigned short sH2[2][16*128];
  __shared__ __attribute__((aligned(16))) unsigned short sRH[2][16*128];
  __shared__ __attribute__((aligned(16))) unsigned short sU [2][16*128];
  __shared__ __attribute__((aligned(16))) float sG[2][16*256];
  __shared__ float sL[256];
  __shared__ float sRed[TPB*2];

  const int t  = threadIdx.x;
  const int b0 = blockIdx.x * 2;

  for (int i = t; i < 256; i += TPB) sL[i] = Lg[i];
  for (int i = t; i < 16*128; i += TPB){
    sH1[0][i] = 0; sH1[1][i] = 0; sH2[0][i] = 0; sH2[1][i] = 0;
  }
  __syncthreads();

  for (int step = 0; step < 5; step++){
    for (int e = 0; e < 2; e++){
      const float* xs = feats + (((size_t)(b0+e))*5 + step)*(16*256);
      for (int i = t; i < 16*256; i += TPB) sX[e][i] = f2bf(xs[i]);
    }
    __syncthreads();

    // layer-1 gates: G = [X|H1] @ W1g  (K=384 -> 12 k-tiles, N=256 -> 16 n-tiles)
    mm2(sX[0], sX[1], 256, 8, sH1[0], sH1[1], 128, 4, pW1g, 16, sG[0], sG[1], 256);
    __syncthreads();
    gates_ew(sL, sG[0], sH1[0], sRH[0], sU[0], b1g);
    gates_ew(sL, sG[1], sH1[1], sRH[1], sU[1], b1g);
    __syncthreads();
    // layer-1 cand: G = [X|r*H1] @ W1c  (N=128)
    mm2(sX[0], sX[1], 256, 8, sRH[0], sRH[1], 128, 4, pW1c, 8, sG[0], sG[1], 128);
    __syncthreads();
    cand_ew(sL, sG[0], sU[0], sH1[0]);
    cand_ew(sL, sG[1], sU[1], sH1[1]);
    __syncthreads();
    // layer-2 gates: G = [H1|H2] @ W2g  (K=256 -> 8 k-tiles)
    mm2(sH1[0], sH1[1], 128, 4, sH2[0], sH2[1], 128, 4, pW2g, 16, sG[0], sG[1], 256);
    __syncthreads();
    gates_ew(sL, sG[0], sH2[0], sRH[0], sU[0], b2g);
    gates_ew(sL, sG[1], sH2[1], sRH[1], sU[1], b2g);
    __syncthreads();
    // layer-2 cand: G = [H1|r*H2] @ W2c
    mm2(sH1[0], sH1[1], 128, 4, sRH[0], sRH[1], 128, 4, pW2c, 8, sG[0], sG[1], 128);
    __syncthreads();
    cand_ew(sL, sG[0], sU[0], sH2[0]);
    cand_ew(sL, sG[1], sU[1], sH2[1]);
    __syncthreads();
  }

  // head: BN(inference) -> ReLU -> Linear(2048 -> 2)
  for (int e = 0; e < 2; e++){
    float a0 = 0.f, a1 = 0.f;
    for (int k = t; k < 2048; k += TPB){
      const float x = bf2f(sH2[e][k]);            // k = a*128 + c  (matches reshape)
      float y = bng[k]*(x - bnm[k])*rsqrtf(bnv[k] + 1e-5f) + bnb[k];
      y = fmaxf(y, 0.f);
      a0 += y * fcw[k*2 + 0];
      a1 += y * fcw[k*2 + 1];
    }
    sRed[t*2] = a0; sRed[t*2 + 1] = a1;
    __syncthreads();
    for (int s = TPB/2; s > 0; s >>= 1){
      if (t < s){ sRed[t*2] += sRed[(t+s)*2]; sRed[t*2+1] += sRed[(t+s)*2+1]; }
      __syncthreads();
    }
    if (t == 0){
      out[(b0+e)*2 + 0] = sRed[0] + fcb[0];
      out[(b0+e)*2 + 1] = sRed[1] + fcb[1];
    }
    __syncthreads();
  }
}

// L = D^-1/2 (A+I) D^-1/2  (T-GCN convention, adj not symmetric: L[i,j]=a[j,i]*d[i]*d[j])
__global__ void lap_kernel(const float* __restrict__ adj, float* __restrict__ L){
  __shared__ float a[256];
  __shared__ float d[16];
  const int t = threadIdx.x;
  if (t < 256) a[t] = adj[t] + (((t >> 4) == (t & 15)) ? 1.f : 0.f);
  __syncthreads();
  if (t < 16){
    float s = 0.f;
    for (int j = 0; j < 16; j++) s += a[t*16 + j];
    d[t] = (s > 0.f) ? rsqrtf(s) : 0.f;
  }
  __syncthreads();
  if (t < 256){
    const int i = t >> 4, j = t & 15;
    L[t] = a[j*16 + i] * d[i] * d[j];
  }
}

// Repack fp32 row-major [K][N] weights into bf16 WMMA-B fragment order:
// tile = nt*KT + kt ; per tile: lane l holds 16 contiguous bf16; elem i -> K = kt*32 + 8*(l>>4) + (i<8 ? i : 16+(i-8)), N = nt*16 + (l&15)
__global__ void repack_kernel(const float* __restrict__ src, unsigned short* __restrict__ dst,
                              int K, int N){
  const int idx = blockIdx.x * 256 + threadIdx.x;
  if (idx >= K*N) return;
  const int i    = idx & 15;
  const int lane = (idx >> 4) & 31;
  const int tile = idx >> 9;
  const int KT   = K >> 5;
  const int kt   = tile % KT;
  const int nt   = tile / KT;
  const int k = kt*32 + ((lane >> 4) << 3) + (i < 8 ? i : 16 + (i - 8));
  const int n = nt*16 + (lane & 15);
  dst[idx] = f2bf(src[k*N + n]);
}

extern "C" void kernel_launch(void* const* d_in, const int* in_sizes, int n_in,
                              void* d_out, int out_size, void* d_ws, size_t ws_size,
                              hipStream_t stream)
{
  const float* feats = (const float*)d_in[0];
  const float* adj   = (const float*)d_in[1];
  const float* w1g   = (const float*)d_in[2];
  const float* b1g   = (const float*)d_in[3];
  const float* w1c   = (const float*)d_in[4];
  const float* w2g   = (const float*)d_in[5];
  const float* b2g   = (const float*)d_in[6];
  const float* w2c   = (const float*)d_in[7];
  const float* bng   = (const float*)d_in[8];
  const float* bnb   = (const float*)d_in[9];
  const float* bnm   = (const float*)d_in[10];
  const float* bnv   = (const float*)d_in[11];
  const float* fcw   = (const float*)d_in[12];
  const float* fcb   = (const float*)d_in[13];

  char* ws = (char*)d_ws;
  float*          L    = (float*)ws;                         // 1 KB
  unsigned short* pW1g = (unsigned short*)(ws + 1024);       // 384*256 bf16 = 192 KB
  unsigned short* pW1c = (unsigned short*)(ws + 197632);     // 384*128 bf16 =  96 KB
  unsigned short* pW2g = (unsigned short*)(ws + 295936);     // 256*256 bf16 = 128 KB
  unsigned short* pW2c = (unsigned short*)(ws + 427008);     // 256*128 bf16 =  64 KB

  lap_kernel<<<1, 256, 0, stream>>>(adj, L);
  repack_kernel<<<(384*256 + 255)/256, 256, 0, stream>>>(w1g, pW1g, 384, 256);
  repack_kernel<<<(384*128 + 255)/256, 256, 0, stream>>>(w1c, pW1c, 384, 128);
  repack_kernel<<<(256*256 + 255)/256, 256, 0, stream>>>(w2g, pW2g, 256, 256);
  repack_kernel<<<(256*128 + 255)/256, 256, 0, stream>>>(w2c, pW2c, 256, 128);

  tgcn_kernel<<<4096/2, TPB, 0, stream>>>(feats, L, pW1g, b1g, pW1c, pW2g, b2g, pW2c,
                                          bng, bnb, bnm, bnv, fcw, fcb, (float*)d_out);
}
